// Model_47407849013794
// MI455X (gfx1250) — matrix operand order repaired and checked by reading, not compiled, not run on hardware
//
#include <hip/hip_runtime.h>

// GRU(B=256,T=2048,I=64,H=32) + pools + FC, fused for MI455X (gfx1250, wave32).
// Serial-latency-bound: 16 single-wave blocks (one 16-row WMMA tile each),
// weights resident as bf16 WMMA B-fragments, x-path pipelined off the h-chain.
// Round 2: hardware bf16 converts via (__bf16) cast; t-loop unroll 2 to kill
// the x double-buffer register copies.

#define TT 2048
#define II 64
#define HH 32

typedef __attribute__((ext_vector_type(16))) __bf16 v16bf;
typedef __attribute__((ext_vector_type(8)))  float  v8f;

union BF16x16 { v16bf v; __bf16 b[16]; unsigned short s[16]; };

__device__ __forceinline__ float fexp2(float x) {
#if __has_builtin(__builtin_amdgcn_exp2f)
  return __builtin_amdgcn_exp2f(x);
#else
  return exp2f(x);
#endif
}
__device__ __forceinline__ float frcp(float x) {
#if __has_builtin(__builtin_amdgcn_rcpf)
  return __builtin_amdgcn_rcpf(x);
#else
  return 1.0f / x;
#endif
}
__device__ __forceinline__ float sigm(float x)   { return frcp(1.0f + fexp2(x * -1.44269504f)); }
__device__ __forceinline__ float tanh_f(float x) { return 1.0f - 2.0f * frcp(fexp2(x * 2.88539008f) + 1.0f); }

// hardware bf16 convert (RNE); backend should pair these into v_cvt_pk_bf16_f32
__device__ __forceinline__ void cvt4(BF16x16& d, int e, float4 a) {
  d.b[e + 0] = (__bf16)a.x; d.b[e + 1] = (__bf16)a.y;
  d.b[e + 2] = (__bf16)a.z; d.b[e + 3] = (__bf16)a.w;
}

#define WMMA_BF16(A, Bm, C) \
  __builtin_amdgcn_wmma_f32_16x16x32_bf16(false, (A), false, (Bm), (short)0, (C), false, false)

__global__ __launch_bounds__(32)
void gru_fused_kernel(const float* __restrict__ x,
                      const float* __restrict__ W_ih,
                      const float* __restrict__ W_hh,
                      const float* __restrict__ b_ih,
                      const float* __restrict__ b_hh,
                      const float* __restrict__ fc_w,
                      const float* __restrict__ fc_b,
                      float* __restrict__ out)
{
  const int lane = threadIdx.x & 31;
  const int nl   = lane & 15;     // column / row-in-tile index
  const int hi   = lane >> 4;     // half-wave select
  const int b0   = blockIdx.x * 16;

  // h staged as f32, row stride 36 floats (144B: 16B aligned, bank-spread, pad)
  __shared__ __align__(16) float hbuf[16 * 36];
  __shared__ float feat[16 * 96];

  // ---------------- Pre-build weight B-fragments (bf16, K x N) ---------------
  // B-frag element e of lane: value = Wt[k = hi*8 + e + (e & 8)][n = tile*16 + nl]
  BF16x16 WI[6][2];   // W_ih^T : 6 N-tiles x 2 K-halves (K=64)
  BF16x16 WH[6];      // W_hh^T : 6 N-tiles (K=32)
  #pragma unroll
  for (int tile = 0; tile < 6; ++tile) {
    const int n = tile * 16 + nl;
    #pragma unroll
    for (int kh = 0; kh < 2; ++kh) {
      const float* p = W_ih + n * II + kh * 32 + hi * 8;
      #pragma unroll
      for (int e = 0; e < 16; ++e) WI[tile][kh].b[e] = (__bf16)p[e + (e & 8)];
    }
    const float* q = W_hh + n * HH + hi * 8;
    #pragma unroll
    for (int e = 0; e < 16; ++e) WH[tile].b[e] = (__bf16)q[e + (e & 8)];
  }

  // Persistent broadcast bias C-operands (no per-step accumulator init cost).
  v8f biasRZ[4], biasXN[2], biasHN[2];
  #pragma unroll
  for (int i = 0; i < 4; ++i) {
    const float bv = b_ih[i * 16 + nl] + b_hh[i * 16 + nl];
    #pragma unroll
    for (int v = 0; v < 8; ++v) biasRZ[i][v] = bv;
  }
  #pragma unroll
  for (int g = 0; g < 2; ++g) {
    const float bx = b_ih[64 + g * 16 + nl];
    const float bh = b_hh[64 + g * 16 + nl];
    #pragma unroll
    for (int v = 0; v < 8; ++v) { biasXN[g][v] = bx; biasHN[g][v] = bh; }
  }

  for (int i = lane; i < 16 * 36; i += 32) hbuf[i] = 0.0f;
  __syncthreads();

  const float* xrow = x + (size_t)(b0 + nl) * (size_t)(TT * II);

  // h / pooling state in C-layout: [g = col-half 0..1][v = 8 rows per half-wave]
  v8f h[2], sum[2], mx[2];
  #pragma unroll
  for (int g = 0; g < 2; ++g)
    #pragma unroll
    for (int v = 0; v < 8; ++v) { h[g][v] = 0.0f; sum[g][v] = 0.0f; mx[g][v] = -3.4e38f; }

  // Preload x A-tile data for t=0 (A-layout addressing: e0-7 -> k=hi*8.., e8-15 -> k=16+hi*8..)
  float4 xc[8];
  {
    const float* p = xrow;
    #pragma unroll
    for (int kh = 0; kh < 2; ++kh) {
      xc[kh * 4 + 0] = *(const float4*)(p + kh * 32 + hi * 8);
      xc[kh * 4 + 1] = *(const float4*)(p + kh * 32 + hi * 8 + 4);
      xc[kh * 4 + 2] = *(const float4*)(p + kh * 32 + 16 + hi * 8);
      xc[kh * 4 + 3] = *(const float4*)(p + kh * 32 + 16 + hi * 8 + 4);
    }
  }

  #pragma unroll 2
  for (int t = 0; t < TT; ++t) {
    // ---- issue next-step x loads + long-range prefetch (independent of h) ----
    const int tn = (t < TT - 1) ? t + 1 : t;
    const float* pn = xrow + (size_t)tn * II;
    float4 xnx[8];
    #pragma unroll
    for (int kh = 0; kh < 2; ++kh) {
      xnx[kh * 4 + 0] = *(const float4*)(pn + kh * 32 + hi * 8);
      xnx[kh * 4 + 1] = *(const float4*)(pn + kh * 32 + hi * 8 + 4);
      xnx[kh * 4 + 2] = *(const float4*)(pn + kh * 32 + 16 + hi * 8);
      xnx[kh * 4 + 3] = *(const float4*)(pn + kh * 32 + 16 + hi * 8 + 4);
    }
    const int tp = (t + 8 < TT) ? t + 8 : TT - 1;
    __builtin_prefetch(xrow + (size_t)tp * II + hi * 32, 0, 3);   // global_prefetch_b8

    // ---- x A-fragments (bf16) for current t ----
    BF16x16 ax0, ax1;
    cvt4(ax0, 0, xc[0]); cvt4(ax0, 4, xc[1]); cvt4(ax0, 8, xc[2]); cvt4(ax0, 12, xc[3]);
    cvt4(ax1, 0, xc[4]); cvt4(ax1, 4, xc[5]); cvt4(ax1, 8, xc[6]); cvt4(ax1, 12, xc[7]);

    // ---- x-side WMMAs: r,z tiles (0..3) and xn tiles (4,5) kept separate ----
    v8f acc[4], xnacc[2], hnacc[2];
    #pragma unroll
    for (int i = 0; i < 4; ++i) {
      acc[i] = WMMA_BF16(ax0.v, WI[i][0].v, biasRZ[i]);
      acc[i] = WMMA_BF16(ax1.v, WI[i][1].v, acc[i]);
    }
    #pragma unroll
    for (int g = 0; g < 2; ++g) {
      xnacc[g] = WMMA_BF16(ax0.v, WI[4 + g][0].v, biasXN[g]);
      xnacc[g] = WMMA_BF16(ax1.v, WI[4 + g][1].v, xnacc[g]);
    }

    // ---- h A-fragment from LDS (C-layout -> A-layout transpose) ----
    BF16x16 hA;
    {
      const float* hr = hbuf + nl * 36;
      float4 h0 = *(const float4*)(hr + hi * 8);
      float4 h1 = *(const float4*)(hr + hi * 8 + 4);
      float4 h2 = *(const float4*)(hr + 16 + hi * 8);
      float4 h3 = *(const float4*)(hr + 16 + hi * 8 + 4);
      cvt4(hA, 0, h0); cvt4(hA, 4, h1); cvt4(hA, 8, h2); cvt4(hA, 12, h3);
    }
    #pragma unroll
    for (int i = 0; i < 4; ++i) acc[i] = WMMA_BF16(hA.v, WH[i].v, acc[i]);
    #pragma unroll
    for (int g = 0; g < 2; ++g)  hnacc[g] = WMMA_BF16(hA.v, WH[4 + g].v, biasHN[g]);

    // ---- gates, state update, pooling, write h back to LDS ----
    #pragma unroll
    for (int g = 0; g < 2; ++g) {
      #pragma unroll
      for (int v = 0; v < 8; ++v) {
        const float r  = sigm(acc[g][v]);
        const float z  = sigm(acc[2 + g][v]);
        const float n  = tanh_f(xnacc[g][v] + r * hnacc[g][v]);
        const float hv = n + z * (h[g][v] - n);     // (1-z)*n + z*h
        h[g][v]   = hv;
        sum[g][v] += hv;
        mx[g][v]  = fmaxf(mx[g][v], hv);
        hbuf[(v + hi * 8) * 36 + g * 16 + nl] = hv; // conflict-free b32 stores
      }
    }

    #pragma unroll
    for (int i = 0; i < 8; ++i) xc[i] = xnx[i];
  }

  // ---------------- Epilogue: feat = [h_last | mean | max], then FC ----------
  const float invT = 1.0f / (float)TT;
  #pragma unroll
  for (int g = 0; g < 2; ++g)
    #pragma unroll
    for (int v = 0; v < 8; ++v) {
      const int row = v + hi * 8;
      const int col = g * 16 + nl;
      feat[row * 96 + col]      = h[g][v];
      feat[row * 96 + 32 + col] = sum[g][v] * invT;
      feat[row * 96 + 64 + col] = mx[g][v];
    }
  __syncthreads();

  if (lane < 16) {
    const int b = b0 + lane;
    #pragma unroll
    for (int o = 0; o < 3; ++o) {
      float s = fc_b[o];
      for (int j = 0; j < 96; ++j) s += feat[lane * 96 + j] * fc_w[o * 96 + j];
      out[b * 3 + o] = s;
    }
  }
}

extern "C" void kernel_launch(void* const* d_in, const int* in_sizes, int n_in,
                              void* d_out, int out_size, void* d_ws, size_t ws_size,
                              hipStream_t stream) {
  (void)in_sizes; (void)n_in; (void)out_size; (void)d_ws; (void)ws_size;
  const float* x    = (const float*)d_in[0];
  const float* W_ih = (const float*)d_in[1];
  const float* W_hh = (const float*)d_in[2];
  const float* b_ih = (const float*)d_in[3];
  const float* b_hh = (const float*)d_in[4];
  const float* fc_w = (const float*)d_in[5];
  const float* fc_b = (const float*)d_in[6];
  gru_fused_kernel<<<16, 32, 0, stream>>>(x, W_ih, W_hh, b_ih, b_hh, fc_w, fc_b,
                                          (float*)d_out);
}
